// RoPEAttention_8186207666943
// MI455X (gfx1250) — compile-verified, matrix operand order checked by reading, not run on hardware
//
#include <hip/hip_runtime.h>
#include <math.h>

typedef __attribute__((ext_vector_type(16))) __bf16 v16bf;
typedef __attribute__((ext_vector_type(8)))  __bf16 v8bf;
typedef __attribute__((ext_vector_type(8)))  float  v8f;

#define WMMA_BF16(a, b, c) \
  __builtin_amdgcn_wmma_f32_16x16x32_bf16(false, (a), false, (b), (short)0, (c), false, false)

static constexpr int Bz = 2;
static constexpr int T  = 2048;
static constexpr int H  = 16;
static constexpr int Dh = 64;
static constexpr int Dm = 1024;

// ---------------------------------------------------------------------------
// fp32 -> bf16 conversion (grid-stride)
// ---------------------------------------------------------------------------
__global__ void f32_to_bf16(const float* __restrict__ src, __bf16* __restrict__ dst, int n) {
  int i = blockIdx.x * blockDim.x + threadIdx.x;
  int stride = gridDim.x * blockDim.x;
  for (; i < n; i += stride) dst[i] = (__bf16)src[i];
}

// ---------------------------------------------------------------------------
// Tiled bf16 WMMA GEMM: C[M,N] = A[M,K] @ B[K,N]
// Block tile 128(M) x 128(N), K-step 32. 256 threads = 8 waves.
// Wave w owns 2 m-tiles ((w&3)*2..+2) x 4 n-tiles ((w>>2)*4..+4):
// 8 WMMAs per K-step from 6 LDS fragment reads.
// A tile (8KB) is staged with GLOBAL_LOAD_ASYNC_TO_LDS_B128 (ASYNCcnt path);
// B tile is register-loaded and stored transposed so both fragment types
// read 16 contiguous K elements per lane.
// ---------------------------------------------------------------------------
template <bool OUT_F32>
__global__ __launch_bounds__(256)
void gemm_bf16_wmma(const __bf16* __restrict__ A, const __bf16* __restrict__ B,
                    void* __restrict__ C, int M, int N, int K) {
  __shared__ __bf16 As[128 * 32];   // 8 KB, [m][k]
  __shared__ __bf16 Bt[128 * 32];   // 8 KB, [n][k]

  const int tid   = threadIdx.x;
  const int wave  = tid >> 5;
  const int lane  = tid & 31;
  const int lrow  = lane & 15;
  const int lhalf = lane >> 4;
  const int m0 = blockIdx.y * 128;
  const int n0 = blockIdx.x * 128;
  const int mt2 = wave & 3;   // pair of m-tiles
  const int ng  = wave >> 2;  // n-half

  // cooperative staging assignments
  const int ar = tid >> 1, ac = (tid & 1) * 16;   // A: 128 rows x 32 cols, 16 elems/thread
  const int bk = tid >> 3, bn = (tid & 7) * 16;   // B: 32 rows x 128 cols, 16 elems/thread

  v8f acc[2][4] = {};

  for (int k0 = 0; k0 < K; k0 += 32) {
    // B slab into registers (no LDS touched yet)
    v16bf bv = *(const v16bf*)(B + (size_t)(k0 + bk) * N + n0 + bn);

    __syncthreads();  // previous iteration done reading LDS

    // A tile: async DMA global -> LDS, 2 x b128 per thread (32 B)
    {
      const __bf16* agp = A + (size_t)(m0 + ar) * K + k0 + ac;
      unsigned lds_off = (unsigned)(uintptr_t)(&As[ar * 32 + ac]);
      asm volatile(
          "global_load_async_to_lds_b128 %0, %1, off\n\t"
          "global_load_async_to_lds_b128 %0, %1, off offset:16"
          :: "v"(lds_off), "v"(agp) : "memory");
    }

    // B tile: transpose into LDS [n][k]
#pragma unroll
    for (int i = 0; i < 16; i++) Bt[(bn + i) * 32 + bk] = bv[i];

    if (k0 + 32 < K)
      __builtin_prefetch(B + (size_t)(k0 + 32 + bk) * N + n0 + bn, 0, 1);

    asm volatile("s_wait_asynccnt 0" ::: "memory");
    __syncthreads();

    v16bf af[2];
#pragma unroll
    for (int mi = 0; mi < 2; mi++)
      af[mi] = *(const v16bf*)(&As[((mt2 * 2 + mi) * 16 + lrow) * 32 + lhalf * 16]);
#pragma unroll
    for (int nt = 0; nt < 4; nt++) {
      v16bf bfrag = *(const v16bf*)(&Bt[((ng * 4 + nt) * 16 + lrow) * 32 + lhalf * 16]);
#pragma unroll
      for (int mi = 0; mi < 2; mi++)
        acc[mi][nt] = WMMA_BF16(af[mi], bfrag, acc[mi][nt]);
    }
  }

  // epilogue: C layout — VGPR i holds row i (lanes 0-15) / row i+8 (lanes 16-31)
#pragma unroll
  for (int mi = 0; mi < 2; mi++) {
#pragma unroll
    for (int nt = 0; nt < 4; nt++) {
      int col = n0 + (ng * 4 + nt) * 16 + lrow;
#pragma unroll
      for (int i = 0; i < 8; i++) {
        int row = m0 + (mt2 * 2 + mi) * 16 + lhalf * 8 + i;
        if (OUT_F32)
          ((float*)C)[(size_t)row * N + col] = acc[mi][nt][i];
        else
          ((__bf16*)C)[(size_t)row * N + col] = (__bf16)acc[mi][nt][i];
      }
    }
  }
}

// ---------------------------------------------------------------------------
// RoPE + relayout: qkv[4096,3072] bf16 -> q,k [B,H,T,D] bf16, v -> vt [B,H,D,T]
// Thread handles one (b,h,t,j) pair covering d=j and d=j+32.
// ---------------------------------------------------------------------------
__global__ void rope_split(const __bf16* __restrict__ qkv, __bf16* __restrict__ q,
                           __bf16* __restrict__ kmat, __bf16* __restrict__ vt) {
  int idx = blockIdx.x * blockDim.x + threadIdx.x;
  if (idx >= Bz * H * T * 32) return;
  int j = idx & 31;
  int t = (idx >> 5) & (T - 1);
  int h = (idx >> 16) & (H - 1);
  int b = idx >> 20;

  size_t row = (size_t)(b * T + t) * (3 * Dm) + (size_t)h * Dh;
  float q1 = (float)qkv[row + j],          q2 = (float)qkv[row + j + 32];
  float k1 = (float)qkv[row + Dm + j],     k2 = (float)qkv[row + Dm + j + 32];
  float v1 = (float)qkv[row + 2 * Dm + j], v2 = (float)qkv[row + 2 * Dm + j + 32];

  // inv_freq = 10000^(-j/32) ; ln(10000)/32 = 0.28782313662425572
  float ang = (float)t * __expf(-(float)j * 0.28782313662425572f);
  float s, c;
  __sincosf(ang, &s, &c);

  int bh = b * H + h;
  size_t qbase = ((size_t)bh * T + t) * Dh;
  q[qbase + j]         = (__bf16)(q1 * c - q2 * s);
  q[qbase + j + 32]    = (__bf16)(q2 * c + q1 * s);
  kmat[qbase + j]      = (__bf16)(k1 * c - k2 * s);
  kmat[qbase + j + 32] = (__bf16)(k2 * c + k1 * s);
  vt[((size_t)bh * Dh + j) * T + t]      = (__bf16)v1;
  vt[((size_t)bh * Dh + j + 32) * T + t] = (__bf16)v2;
}

// ---------------------------------------------------------------------------
// Flash attention, non-causal. Grid (T/64, H, B), 128 threads = 4 waves.
// Each wave owns 16 query rows; key blocks of 64; online softmax; P goes
// through a per-wave LDS tile to convert C-layout -> A-layout for P@V.
// Output written as [B,T,H,D] bf16 (== [4096,1024] for the final GEMM).
// ---------------------------------------------------------------------------
__global__ __launch_bounds__(128)
void flash_attn(const __bf16* __restrict__ q, const __bf16* __restrict__ k,
                const __bf16* __restrict__ vt, __bf16* __restrict__ o) {
  __shared__ __bf16 pbuf[4][16 * 64];  // per-wave 2 KB P tile

  const int lane  = threadIdx.x & 31;
  const int wave  = threadIdx.x >> 5;
  const int lrow  = lane & 15;
  const int lhalf = lane >> 4;
  const int h  = blockIdx.y;
  const int b  = blockIdx.z;
  const int bh = b * H + h;
  const int m0 = blockIdx.x * 64 + wave * 16;

  const size_t qkbase = (size_t)bh * T * Dh;
  const size_t vbase  = (size_t)bh * Dh * T;

  // Q fragments (16 rows x 64 d -> two 16x32 A-frags), loaded once
  v16bf aq[2];
#pragma unroll
  for (int ks = 0; ks < 2; ks++)
    aq[ks] = *(const v16bf*)(q + qkbase + (size_t)(m0 + lrow) * Dh + ks * 32 + lhalf * 16);

  v8f oacc[4] = {};
  float mrow[8], lsum[8];
#pragma unroll
  for (int i = 0; i < 8; i++) { mrow[i] = -1e30f; lsum[i] = 0.f; }

  for (int j0 = 0; j0 < T; j0 += 64) {
    // S = (Q K^T) * scale : 16 x 64 in 4 C-tiles
    v8f sacc[4];
#pragma unroll
    for (int nt = 0; nt < 4; nt++) {
      v8f s = {};
#pragma unroll
      for (int ks = 0; ks < 2; ks++) {
        v16bf bk = *(const v16bf*)(k + qkbase + (size_t)(j0 + nt * 16 + lrow) * Dh +
                                   ks * 32 + lhalf * 16);
        s = WMMA_BF16(aq[ks], bk, s);
      }
#pragma unroll
      for (int i = 0; i < 8; i++) s[i] *= 0.125f;  // 1/sqrt(64)
      sacc[nt] = s;
    }

    // online softmax row statistics (row i lives in VGPR i across a 16-lane half)
    float mnew[8], corr[8], psum[8];
#pragma unroll
    for (int i = 0; i < 8; i++) {
      float mx = fmaxf(fmaxf(sacc[0][i], sacc[1][i]), fmaxf(sacc[2][i], sacc[3][i]));
#pragma unroll
      for (int msk = 1; msk < 16; msk <<= 1) mx = fmaxf(mx, __shfl_xor(mx, msk, 32));
      mnew[i] = fmaxf(mrow[i], mx);
      corr[i] = __expf(mrow[i] - mnew[i]);
      mrow[i] = mnew[i];
      psum[i] = 0.f;
    }

    // P = exp(S - m), spill to LDS in row-major for A-layout reload
#pragma unroll
    for (int nt = 0; nt < 4; nt++) {
#pragma unroll
      for (int i = 0; i < 8; i++) {
        float p = __expf(sacc[nt][i] - mnew[i]);
        psum[i] += p;
        pbuf[wave][(lhalf * 8 + i) * 64 + nt * 16 + lrow] = (__bf16)p;
      }
    }
#pragma unroll
    for (int i = 0; i < 8; i++) {
      float ps = psum[i];
#pragma unroll
      for (int msk = 1; msk < 16; msk <<= 1) ps += __shfl_xor(ps, msk, 32);
      lsum[i] = lsum[i] * corr[i] + ps;
    }
#pragma unroll
    for (int dt = 0; dt < 4; dt++)
#pragma unroll
      for (int i = 0; i < 8; i++) oacc[dt][i] *= corr[i];

    // O += P @ V  (K-dim = 64 keys; V transposed so B-frags are contiguous in T)
    v16bf pf[2];
#pragma unroll
    for (int ks = 0; ks < 2; ks++)
      pf[ks] = *(const v16bf*)(&pbuf[wave][lrow * 64 + ks * 32 + lhalf * 16]);
#pragma unroll
    for (int dt = 0; dt < 4; dt++) {
#pragma unroll
      for (int ks = 0; ks < 2; ks++) {
        v16bf bv = *(const v16bf*)(vt + vbase + (size_t)(dt * 16 + lrow) * T +
                                   j0 + ks * 32 + lhalf * 16);
        oacc[dt] = WMMA_BF16(pf[ks], bv, oacc[dt]);
      }
    }
  }

  // normalize and write [B,T,H,D]
#pragma unroll
  for (int dt = 0; dt < 4; dt++) {
#pragma unroll
    for (int i = 0; i < 8; i++) {
      int t = m0 + lhalf * 8 + i;
      float val = oacc[dt][i] / lsum[i];
      o[((size_t)(b * T + t) * H + h) * Dh + dt * 16 + lrow] = (__bf16)val;
    }
  }
}

// ---------------------------------------------------------------------------
// launch
// ---------------------------------------------------------------------------
extern "C" void kernel_launch(void* const* d_in, const int* in_sizes, int n_in,
                              void* d_out, int out_size, void* d_ws, size_t ws_size,
                              hipStream_t stream) {
  const float* x     = (const float*)d_in[0];
  const float* w_qkv = (const float*)d_in[1];
  const float* w_o   = (const float*)d_in[2];

  char* ws = (char*)d_ws;
  size_t off = 0;
  auto alloc = [&](size_t bytes) -> void* {
    void* p = ws + off;
    off += (bytes + 255) & ~(size_t)255;
    return p;
  };

  const size_t MT = (size_t)Bz * T;  // 4096
  __bf16* xb   = (__bf16*)alloc(MT * Dm * 2);
  __bf16* wqb  = (__bf16*)alloc((size_t)Dm * 3 * Dm * 2);
  __bf16* wob  = (__bf16*)alloc((size_t)Dm * Dm * 2);
  __bf16* qkvb = (__bf16*)alloc(MT * 3 * Dm * 2);
  __bf16* qb   = (__bf16*)alloc((size_t)Bz * H * T * Dh * 2);
  __bf16* kb   = (__bf16*)alloc((size_t)Bz * H * T * Dh * 2);
  __bf16* vtb  = (__bf16*)alloc((size_t)Bz * H * Dh * T * 2);
  __bf16* ab   = (__bf16*)alloc(MT * Dm * 2);

  f32_to_bf16<<<8192, 256, 0, stream>>>(x, xb, (int)(MT * Dm));
  f32_to_bf16<<<8192, 256, 0, stream>>>(w_qkv, wqb, Dm * 3 * Dm);
  f32_to_bf16<<<4096, 256, 0, stream>>>(w_o, wob, Dm * Dm);

  // qkv = x @ w_qkv  (M=4096, N=3072, K=1024)
  gemm_bf16_wmma<false><<<dim3(3 * Dm / 128, MT / 128), 256, 0, stream>>>(
      xb, wqb, qkvb, (int)MT, 3 * Dm, Dm);

  // RoPE + relayout
  rope_split<<<(Bz * H * T * 32) / 256, 256, 0, stream>>>(qkvb, qb, kb, vtb);

  // attention
  flash_attn<<<dim3(T / 64, H, Bz), 128, 0, stream>>>(qb, kb, vtb, ab);

  // out = attn @ w_o  (M=4096, N=1024, K=1024), fp32 epilogue
  gemm_bf16_wmma<true><<<dim3(Dm / 128, MT / 128), 256, 0, stream>>>(
      ab, wob, d_out, (int)MT, Dm, Dm);
}
